// SigKAN_34926674051270
// MI455X (gfx1250) — matrix operand order repaired and checked by reading, not compiled, not run on hardware
//
#include <hip/hip_runtime.h>
#include <hip/hip_bf16.h>

typedef __attribute__((ext_vector_type(16))) __bf16 v16bf;
typedef __attribute__((ext_vector_type(8)))  float  v8f;

#define BATCH 256
#define SEQ   512
#define FEAT  64
#define NC    8
#define KTOT  576          // 64 silu-base + 64*8 spline features
#define WSTR  292          // u32 stride of one weight row in LDS (584 bf16, padded from 576)
#define XSTR  68           // float stride of one x row in LDS (16B-aligned, bank-safe)

// fast silu: x * rcp(1+exp(-x)); v_rcp_f32 is ~1ulp, far below bf16 rounding
static __device__ __forceinline__ float siluf(float x) {
  return x * __builtin_amdgcn_rcpf(1.0f + __expf(-x));
}

static __device__ __forceinline__ __bf16 tobf(float f) { return (__bf16)f; }

// 8 cubic B-spline bases (Cox-de Boor), uniform grid g[j] = (j-3)*0.4 - 1, j=0..11
static __device__ __forceinline__ void bspline8(float x, float* o) {
  const float h = 0.4f;
  float g[12];
#pragma unroll
  for (int j = 0; j < 12; ++j) g[j] = (float)(j - 3) * h - 1.0f;
  float bb[11];
#pragma unroll
  for (int j = 0; j < 11; ++j) bb[j] = (x >= g[j] && x < g[j + 1]) ? 1.0f : 0.0f;
#pragma unroll
  for (int p = 1; p <= 3; ++p) {
    float inv = 1.0f / (h * (float)p);   // compile-time constant
#pragma unroll
    for (int j = 0; j < 11 - p; ++j)
      bb[j] = (x - g[j]) * inv * bb[j] + (g[j + p + 1] - x) * inv * bb[j + 1];
  }
#pragma unroll
  for (int j = 0; j < 8; ++j) o[j] = bb[j];
}

static __device__ __forceinline__ v16bf load_bfrag(const unsigned int* p) {
  union { v16bf v; unsigned int u[8]; } r;
#pragma unroll
  for (int q = 0; q < 8; ++q) r.u[q] = p[q];
  return r.v;
}

// ---------------------------------------------------------------------------
// Pack combined weight matrix W[k][n] -> bf16, stored transposed as Wt[n][k].
// k<64: wb[n][k] (silu base path);  k>=64: wspl[n][(k-64)/8][(k-64)%8].
// ---------------------------------------------------------------------------
__global__ void wprep_kernel(const float* __restrict__ wb,
                             const float* __restrict__ wspl,
                             unsigned short* __restrict__ Wt) {
  int idx = blockIdx.x * 256 + threadIdx.x;
  if (idx >= 64 * KTOT) return;
  int n = idx / KTOT, k = idx - n * KTOT;
  float v = (k < 64) ? wb[n * 64 + k] : wspl[n * 512 + (k - 64)];
  __bf16 hb = (__bf16)v;
  Wt[idx] = *(unsigned short*)&hb;
}

// ---------------------------------------------------------------------------
// GRKAN gate path: sig -> KAN -> silu -> KAN -> GLU gate -> res LN -> softmax.
// One 64-thread block per batch element. Output: wts[b][o] (fp32).
// ---------------------------------------------------------------------------
__global__ __launch_bounds__(64) void gate_kernel(
    const float* __restrict__ x,
    const float* __restrict__ gwb1, const float* __restrict__ gws1,
    const float* __restrict__ gwb2, const float* __restrict__ gws2,
    const float* __restrict__ wgate, const float* __restrict__ bgate,
    const float* __restrict__ wval,  const float* __restrict__ bval,
    const float* __restrict__ gamma, const float* __restrict__ beta,
    float* __restrict__ wts) {
  const int b = blockIdx.x, o = threadIdx.x;
  __shared__ float sigs[64], feat[64], bs[64 * 8], hbuf[64], red[64];
  const float* xb = x + (long long)b * SEQ * FEAT;
  // log-signature level-1 (level-2 terms are sliced away since UNIT == FEAT)
  float sv = xb[(SEQ - 1) * FEAT + o] - xb[o];
  sigs[o] = sv;
  feat[o] = siluf(sv);
  bspline8(sv, &bs[o * 8]);
  __syncthreads();
  // KAN layer 1
  float acc = 0.f;
  for (int f = 0; f < 64; ++f) {
    acc += feat[f] * gwb1[o * 64 + f];
#pragma unroll
    for (int c = 0; c < 8; ++c) acc += bs[f * 8 + c] * gws1[o * 512 + f * 8 + c];
  }
  float h1 = siluf(acc);          // h after inter-layer silu
  __syncthreads();
  feat[o] = siluf(h1);
  bspline8(h1, &bs[o * 8]);
  __syncthreads();
  // KAN layer 2
  acc = 0.f;
  for (int f = 0; f < 64; ++f) {
    acc += feat[f] * gwb2[o * 64 + f];
#pragma unroll
    for (int c = 0; c < 8; ++c) acc += bs[f * 8 + c] * gws2[o * 512 + f * 8 + c];
  }
  hbuf[o] = acc;
  __syncthreads();
  // GLU gate + value
  float g = bgate[o], v = bval[o];
  for (int j = 0; j < 64; ++j) {
    g += hbuf[j] * wgate[j * 64 + o];
    v += hbuf[j] * wval[j * 64 + o];
  }
  g = __builtin_amdgcn_rcpf(1.0f + __expf(-g));
  float res = sigs[o] + g * v;
  red[o] = res;
  __syncthreads();
  float mu = 0.f, m2 = 0.f;
  for (int j = 0; j < 64; ++j) { mu += red[j]; m2 += red[j] * red[j]; }
  mu *= (1.0f / 64.0f); m2 *= (1.0f / 64.0f);
  float var = m2 - mu * mu;
  float ln = (res - mu) * rsqrtf(var + 1e-5f) * gamma[o] + beta[o];
  __syncthreads();
  red[o] = ln;
  __syncthreads();
  float mx = -3.4e38f;
  for (int j = 0; j < 64; ++j) mx = fmaxf(mx, red[j]);
  float ev = __expf(ln - mx);
  __syncthreads();
  red[o] = ev;
  __syncthreads();
  float sum = 0.f;
  for (int j = 0; j < 64; ++j) sum += red[j];
  wts[b * 64 + o] = ev * __builtin_amdgcn_rcpf(sum);
}

// ---------------------------------------------------------------------------
// Main fused kernel: per row, features = [silu(x[0..63]) | bspline8(x[i])_c],
// out[row] = (features @ W) * weights[b], via v_wmma_f32_16x16x32_bf16.
// 256 threads = 8 waves; wave w computes rows [w*16, w*16+16) x all 64 cols.
// ---------------------------------------------------------------------------
__global__ __launch_bounds__(256) void kanout_kernel(
    const float* __restrict__ x, const unsigned short* __restrict__ Wt,
    const float* __restrict__ wts, float* __restrict__ out) {
  __shared__ __align__(16) float xs[128 * XSTR];        // 128 rows x 64 feats
  __shared__ __align__(16) unsigned int wl[64 * WSTR];  // 64 rows x 584 bf16
  __shared__ float wrow[64];
  const int tid = threadIdx.x;
  const long long row0 = (long long)blockIdx.x * 128;
  const int b = blockIdx.x >> 2;            // 128-row tile never crosses a batch

  if (tid < 64) wrow[tid] = wts[b * 64 + tid];
  // stage weights: 64 x 288 u32 as uint4 (global_load_b128 / ds_store_b128)
  {
    const uint4* ws4 = (const uint4*)Wt;
#pragma unroll 2
    for (int i = tid; i < 64 * 72; i += 256) {
      int n = i / 72, kk = i - n * 72;
      *(uint4*)&wl[n * WSTR + kk * 4] = ws4[i];
    }
  }
  // stage x tile: 128 x 64 floats as float4
  {
    const float4* xsrc = (const float4*)(x + row0 * 64);
#pragma unroll 2
    for (int i = tid; i < 128 * 16; i += 256) {
      int r = i >> 4, f4 = i & 15;
      *(float4*)&xs[r * XSTR + f4 * 4] = xsrc[i];
    }
  }
  __syncthreads();

  const int wv = tid >> 5;                  // wave id 0..7
  const int lane = tid & 31;
  const int mrow = lane & 15;               // A-row / B-col within tile
  const int hi = lane >> 4;                 // lane half (K-group selector)
  const float* xrow = &xs[(wv * 16 + mrow) * XSTR];

  const v8f vzero = {0.f, 0.f, 0.f, 0.f, 0.f, 0.f, 0.f, 0.f};
  v8f acc[4];
#pragma unroll
  for (int t = 0; t < 4; ++t) acc[t] = vzero;

  // --- base (silu) region: K = 0..63, steps s=0,1 (uniform branch structure) ---
#pragma unroll
  for (int s = 0; s < 2; ++s) {
    const int kbase = s * 32;
    v16bf a;
#pragma unroll
    for (int e = 0; e < 16; ++e) {
      int K = kbase + hi * 16 + e;   // contiguous K order, matches B fragment
      a[e] = tobf(siluf(xrow[K]));
    }
#pragma unroll
    for (int t = 0; t < 4; ++t) {
      const unsigned int* bp = &wl[(t * 16 + mrow) * WSTR + (kbase >> 1) + hi * 8];
      v16bf bfr = load_bfrag(bp);
      acc[t] = __builtin_amdgcn_wmma_f32_16x16x32_bf16(false, a, false, bfr,
                                                       (short)0, acc[t], false, false);
    }
  }
  // --- spline region: K = 64..575, steps s=2..17; each aligned 8-block of K is
  //     exactly one feature's 8 Cox-de-Boor coefficients (computed once) ---
#pragma unroll 2
  for (int s = 2; s < 18; ++s) {
    const int fb = 4 * (s - 2);
    float c0[8], c1[8];
    bspline8(xrow[fb + 2 * hi], c0);
    bspline8(xrow[fb + 2 * hi + 1], c1);
    v16bf a;
#pragma unroll
    for (int e = 0; e < 8; ++e) { a[e] = tobf(c0[e]); a[e + 8] = tobf(c1[e]); }
    const int kbase = 32 * s;
#pragma unroll
    for (int t = 0; t < 4; ++t) {
      const unsigned int* bp = &wl[(t * 16 + mrow) * WSTR + (kbase >> 1) + hi * 8];
      v16bf bfr = load_bfrag(bp);
      acc[t] = __builtin_amdgcn_wmma_f32_16x16x32_bf16(false, a, false, bfr,
                                                       (short)0, acc[t], false, false);
    }
  }
  // --- epilogue: D layout (VGPR e -> M = e + hi*8, N = tile*16 + mrow) ---
#pragma unroll
  for (int t = 0; t < 4; ++t) {
    const int n = t * 16 + mrow;
    const float wscale = wrow[n];
#pragma unroll
    for (int e = 0; e < 8; ++e) {
      long long r = row0 + wv * 16 + hi * 8 + e;
      out[r * 64 + n] = acc[t][e] * wscale;
    }
  }
}

extern "C" void kernel_launch(void* const* d_in, const int* in_sizes, int n_in,
                              void* d_out, int out_size, void* d_ws, size_t ws_size,
                              hipStream_t stream) {
  const float* x     = (const float*)d_in[0];
  const float* wb    = (const float*)d_in[1];
  const float* wspl  = (const float*)d_in[2];
  const float* gwb1  = (const float*)d_in[3];
  const float* gws1  = (const float*)d_in[4];
  const float* gwb2  = (const float*)d_in[5];
  const float* gws2  = (const float*)d_in[6];
  const float* wgate = (const float*)d_in[7];
  const float* bgate = (const float*)d_in[8];
  const float* wval  = (const float*)d_in[9];
  const float* bval  = (const float*)d_in[10];
  const float* gamma = (const float*)d_in[11];
  const float* beta  = (const float*)d_in[12];
  float* out = (float*)d_out;

  unsigned short* Wt = (unsigned short*)d_ws;                       // 64*576 bf16
  float* wts = (float*)((char*)d_ws + (size_t)64 * KTOT * sizeof(unsigned short));

  wprep_kernel<<<(64 * KTOT + 255) / 256, 256, 0, stream>>>(wb, wspl, Wt);
  gate_kernel<<<BATCH, 64, 0, stream>>>(x, gwb1, gws1, gwb2, gws2,
                                        wgate, bgate, wval, bval, gamma, beta, wts);
  kanout_kernel<<<(BATCH * SEQ) / 128, 256, 0, stream>>>(x, Wt, wts, out);
}
